// Hyperpixel_MMPN_54073638257172
// MI455X (gfx1250) — matrix operand order repaired and checked by reading, not compiled, not run on hardware
//
#include <hip/hip_runtime.h>
#include <hip/hip_bf16.h>

#define N_PIX 65536
#define N_SUP 4096
#define DD    256
#define E_REG 131072
#define E_IMP 20480

typedef __attribute__((ext_vector_type(16))) __bf16 v16bf;
typedef __attribute__((ext_vector_type(8)))  __bf16 v8bf;
typedef __attribute__((ext_vector_type(8)))  float  v8f;
typedef __attribute__((ext_vector_type(4)))  float  v4f;

// fp32 -> bf16 round-to-nearest-even, bit-level
__device__ __forceinline__ __bf16 f2bf(float f) {
    union { float f; unsigned u; } in; in.f = f;
    unsigned u = in.u;
    unsigned r = (u + 0x7FFFu + ((u >> 16) & 1u)) >> 16;
    union { unsigned short s; __bf16 b; } out; out.s = (unsigned short)r;
    return out.b;
}

__device__ __forceinline__ v16bf combine16(v8bf lo, v8bf hi) {
    v16bf r;
#pragma unroll
    for (int i = 0; i < 8; ++i) { r[i] = lo[i]; r[8 + i] = hi[i]; }
    return r;
}

// ---------------------------------------------------------------------------
// Tiled transpose + fp32 -> bf16:  in[R][256] fp32  ->  out[256][R] bf16
// Grid: (R/64, 4), 256 threads.
// ---------------------------------------------------------------------------
__global__ void __launch_bounds__(256)
transpose_to_bf16(const float* __restrict__ in, __bf16* __restrict__ out, int R)
{
    __shared__ __align__(16) __bf16 tile[64][72];   // [c][r], padded rows (144B = 9*16)
    const int r0 = blockIdx.x * 64;
    const int c0 = blockIdx.y * 64;
    const int t  = threadIdx.x;
    {
        const int rr = t >> 2;              // 0..63
        const int cb = (t & 3) * 16;        // 0..48
        const float* src = &in[(size_t)(r0 + rr) * DD + c0 + cb];
#pragma unroll
        for (int i = 0; i < 4; ++i) {
            const v4f f = *(const v4f*)(src + i * 4);
            tile[cb + i * 4 + 0][rr] = f2bf(f.x);
            tile[cb + i * 4 + 1][rr] = f2bf(f.y);
            tile[cb + i * 4 + 2][rr] = f2bf(f.z);
            tile[cb + i * 4 + 3][rr] = f2bf(f.w);
        }
    }
    __syncthreads();
    {
        const int cc = t >> 2;              // 0..63
        const int rb = (t & 3) * 16;        // 0..48
        __bf16* dst = &out[(size_t)(c0 + cc) * R + r0 + rb];
        *(v8bf*)dst       = *(const v8bf*)&tile[cc][rb];
        *((v8bf*)dst + 1) = *(const v8bf*)&tile[cc][rb + 8];
    }
}

// ---------------------------------------------------------------------------
// C[M x 256] (+)= op(A)[M x K] @ Bt^T, with Bt pre-transposed bf16 [256][K_total].
// TRANS_A=false : A row-major [M][lda]
// TRANS_A=true  : A stored [K][lda] (i.e. op(A) = A^T)
// Block: 256 threads = 8 waves, tile 64(M) x 256(N), K-step 32, split-K via
// blockIdx.y (klen per block); ATOMIC_OUT accumulates C with fp32 atomics.
// A staged double-buffered in LDS (fp32->bf16 on the fly); B fragments are
// direct global bf16 loads (L2-resident), batched 4 tiles at a time so the
// scheduler can use partial s_wait_loadcnt.
// ---------------------------------------------------------------------------
template <bool TRANS_A, bool ATOMIC_OUT>
__global__ void __launch_bounds__(256)
wmma_gemm_bf16(const float* __restrict__ A, const __bf16* __restrict__ Bt,
               float* __restrict__ C, int K_total, int lda, int klen)
{
    __shared__ __align__(16) __bf16 Alds[2][64][32];

    const int tid  = threadIdx.x;
    const int lane = tid & 31;
    const int wave = tid >> 5;
    const int wm   = wave & 3;    // 0..3 -> m-tile
    const int wn   = wave >> 2;   // 0..1 -> 128-col slab
    const int r    = lane & 15;
    const int grp  = lane >> 4;
    const int bm   = blockIdx.x * 64;
    const int k0b  = blockIdx.y * klen;
    const int nk   = klen / 32;

    // staging indices
    const int s_kk  = TRANS_A ? (tid >> 3) : 0;        // TRANS: k row 0..31
    const int s_mm8 = TRANS_A ? ((tid & 7) * 8) : 0;   // TRANS: m base 0..56
    const int s_mm  = TRANS_A ? 0 : (tid >> 2);        // N: m row 0..63
    const int s_kk8 = TRANS_A ? 0 : ((tid & 3) * 8);   // N: k base 0..24

    v8f acc[8] = {};

    // ---- prologue: load + stage tile 0 ----
    v4f f0, f1;
    {
        const int k0 = k0b;
        if (TRANS_A) {
            const float* p = &A[(size_t)(k0 + s_kk) * lda + bm + s_mm8];
            f0 = __builtin_nontemporal_load((const v4f*)p);
            f1 = __builtin_nontemporal_load((const v4f*)(p + 4));
        } else {
            const float* p = &A[(size_t)(bm + s_mm) * lda + k0 + s_kk8];
            f0 = __builtin_nontemporal_load((const v4f*)p);
            f1 = __builtin_nontemporal_load((const v4f*)(p + 4));
        }
    }
    {
        const float v[8] = {f0.x, f0.y, f0.z, f0.w, f1.x, f1.y, f1.z, f1.w};
        if (TRANS_A) {
#pragma unroll
            for (int i = 0; i < 8; ++i) Alds[0][s_mm8 + i][s_kk] = f2bf(v[i]);
        } else {
#pragma unroll
            for (int i = 0; i < 8; ++i) Alds[0][s_mm][s_kk8 + i] = f2bf(v[i]);
        }
    }
    __syncthreads();

    int buf = 0;
    for (int it = 0; it < nk; ++it) {
        const int k0 = k0b + it * 32;
        const bool more = (it + 1) < nk;

        // ---- prefetch next A tile into registers ----
        if (more) {
            const int kn = k0 + 32;
            if (TRANS_A) {
                const float* p = &A[(size_t)(kn + s_kk) * lda + bm + s_mm8];
                f0 = __builtin_nontemporal_load((const v4f*)p);
                f1 = __builtin_nontemporal_load((const v4f*)(p + 4));
            } else {
                const float* p = &A[(size_t)(bm + s_mm) * lda + kn + s_kk8];
                f0 = __builtin_nontemporal_load((const v4f*)p);
                f1 = __builtin_nontemporal_load((const v4f*)(p + 4));
            }
        }

        // ---- A fragment from LDS ----
        const v8bf a_lo = *(const v8bf*)&Alds[buf][wm * 16 + r][grp * 8];
        const v8bf a_hi = *(const v8bf*)&Alds[buf][wm * 16 + r][16 + grp * 8];
        const v16bf afrag = combine16(a_lo, a_hi);

        // ---- B fragments: batches of 4 tiles -> 8 clause'd loads, then 4 WMMAs ----
        const size_t bko = (size_t)k0 + grp * 8;
#pragma unroll
        for (int half = 0; half < 2; ++half) {
            v8bf bfl[4], bfh[4];
#pragma unroll
            for (int q = 0; q < 4; ++q) {
                const int t = half * 4 + q;
                const v8bf* bp =
                    (const v8bf*)&Bt[(size_t)(wn * 128 + t * 16 + r) * K_total + bko];
                bfl[q] = bp[0];
                bfh[q] = bp[2];
            }
#pragma unroll
            for (int q = 0; q < 4; ++q) {
                const int t = half * 4 + q;
                acc[t] = __builtin_amdgcn_wmma_f32_16x16x32_bf16(
                    false, afrag, false, combine16(bfl[q], bfh[q]),
                    (short)0, acc[t], false, false);
            }
        }

        // ---- stage next tile into the other buffer ----
        if (more) {
            const float v[8] = {f0.x, f0.y, f0.z, f0.w, f1.x, f1.y, f1.z, f1.w};
            if (TRANS_A) {
#pragma unroll
                for (int i = 0; i < 8; ++i) Alds[buf ^ 1][s_mm8 + i][s_kk] = f2bf(v[i]);
            } else {
#pragma unroll
                for (int i = 0; i < 8; ++i) Alds[buf ^ 1][s_mm][s_kk8 + i] = f2bf(v[i]);
            }
        }
        __syncthreads();
        buf ^= 1;
    }

    // ---- store / accumulate C ----
#pragma unroll
    for (int t = 0; t < 8; ++t) {
        const int col = wn * 128 + t * 16 + r;
#pragma unroll
        for (int j = 0; j < 8; ++j) {
            const int row = bm + wm * 16 + j + 8 * grp;
            float* dst = &C[(size_t)row * DD + col];
            if (ATOMIC_OUT) unsafeAtomicAdd(dst, acc[t][j]);
            else            __builtin_nontemporal_store(acc[t][j], dst);
        }
    }
}

// ---------------------------------------------------------------------------
__global__ void __launch_bounds__(256)
colsum_kernel(const float* __restrict__ Q, float* __restrict__ colsum)
{
    const int col = blockIdx.x * 256 + threadIdx.x;
    const int r0  = blockIdx.y * 1024;
    float s = 0.0f;
    for (int i = 0; i < 1024; ++i)
        s += Q[(size_t)(r0 + i) * N_SUP + col];
    unsafeAtomicAdd(&colsum[col], s);
}

__global__ void __launch_bounds__(256)
recip_kernel(float* __restrict__ v)
{
    const int i = blockIdx.x * 256 + threadIdx.x;
    v[i] = 1.0f / v[i];
}

__global__ void __launch_bounds__(256)
scale_rows_kernel(float* __restrict__ m, const float* __restrict__ rs)
{
    const int row = blockIdx.x;
    m[(size_t)row * DD + threadIdx.x] *= rs[row];
}

// out[s][d] = sum_k in[s][k] * W[k][d] + b[d]
__global__ void __launch_bounds__(256)
linear_kernel(const float* __restrict__ in, const float* __restrict__ W,
              const float* __restrict__ b, float* __restrict__ out)
{
    __shared__ float row[DD];
    const int s = blockIdx.x;
    const int d = threadIdx.x;
    row[d] = in[(size_t)s * DD + d];
    __syncthreads();
    float acc = b[d];
#pragma unroll 8
    for (int k = 0; k < DD; ++k)
        acc += row[k] * W[(size_t)k * DD + d];
    out[(size_t)s * DD + d] = acc;
}

// agg[dst[e]][d] += w[e] * h[src[e]][d]
__global__ void __launch_bounds__(256)
scatter_kernel(const int* __restrict__ src, const int* __restrict__ dst,
               const float* __restrict__ w, const float* __restrict__ h,
               float* __restrict__ agg)
{
    const int e = blockIdx.x;
    const int d = threadIdx.x;
    const int s = src[e];
    const int t = dst[e];
    const float ww = w[e];
    unsafeAtomicAdd(&agg[(size_t)t * DD + d], ww * h[(size_t)s * DD + d]);
}

__global__ void __launch_bounds__(256)
bn_stats_kernel(const float* __restrict__ agg, float* __restrict__ gsum,
                float* __restrict__ gsumsq)
{
    const int d  = threadIdx.x;
    const int r0 = blockIdx.x * 256;
    float s = 0.0f, s2 = 0.0f;
    for (int i = 0; i < 256; ++i) {
        const float v = agg[(size_t)(r0 + i) * DD + d];
        s += v; s2 += v * v;
    }
    unsafeAtomicAdd(&gsum[d], s);
    unsafeAtomicAdd(&gsumsq[d], s2);
}

__global__ void __launch_bounds__(256)
bn_apply_kernel(const float* __restrict__ agg, const float* __restrict__ gsum,
                const float* __restrict__ gsumsq, const float* __restrict__ g,
                const float* __restrict__ be, float* __restrict__ out)
{
    const int d   = threadIdx.x;
    const size_t idx = (size_t)blockIdx.x * DD + d;
    const float mu  = gsum[d] * (1.0f / N_SUP);
    const float var = gsumsq[d] * (1.0f / N_SUP) - mu * mu;
    const float v = (agg[idx] - mu) * rsqrtf(var + 1e-5f) * g[d] + be[d];
    out[idx] = v >= 0.0f ? v : 0.01f * v;
}

// ---------------------------------------------------------------------------
extern "C" void kernel_launch(void* const* d_in, const int* in_sizes, int n_in,
                              void* d_out, int out_size, void* d_ws, size_t ws_size,
                              hipStream_t stream)
{
    (void)in_sizes; (void)n_in; (void)out_size; (void)ws_size;

    const float* x   = (const float*)d_in[0];
    const float* Q   = (const float*)d_in[1];
    const int*   es  = (const int*)  d_in[2];
    const int*   ed  = (const int*)  d_in[3];
    const float* ew  = (const float*)d_in[4];
    const int*   is  = (const int*)  d_in[5];
    const int*   id  = (const int*)  d_in[6];
    const float* iw  = (const float*)d_in[7];
    const float* W1  = (const float*)d_in[8];
    const float* b1  = (const float*)d_in[9];
    const float* g1  = (const float*)d_in[10];
    const float* be1 = (const float*)d_in[11];
    const float* W2  = (const float*)d_in[12];
    const float* b2  = (const float*)d_in[13];
    const float* g2  = (const float*)d_in[14];
    const float* be2 = (const float*)d_in[15];
    float* out = (float*)d_out;

    const size_t SD = (size_t)N_SUP * DD;
    float* ws     = (float*)d_ws;
    float* rs     = ws;             // N_SUP
    float* hp     = rs + N_SUP;     // SD
    float* lin    = hp + SD;        // SD
    float* agg    = lin + SD;       // SD
    float* h1     = agg + SD;       // SD
    float* h2     = h1 + SD;        // SD
    float* gsum   = h2 + SD;        // DD
    float* gsumsq = gsum + DD;      // DD
    __bf16* xT    = (__bf16*)(gsumsq + DD);          // [256][65536] bf16 (32 MB)
    __bf16* h2T   = xT + (size_t)DD * N_PIX;         // [256][4096]  bf16 (2 MB)

    // 0) xT = bf16(x^T)
    transpose_to_bf16<<<dim3(N_PIX / 64, DD / 64), 256, 0, stream>>>(x, xT, N_PIX);

    // 1) rs = 1 / colsum(Q)
    (void)hipMemsetAsync(rs, 0, N_SUP * sizeof(float), stream);
    colsum_kernel<<<dim3(N_SUP / 256, N_PIX / 1024), 256, 0, stream>>>(Q, rs);
    recip_kernel<<<N_SUP / 256, 256, 0, stream>>>(rs);

    // 2) hp = (Q^T @ x), split-K x8 with atomic accumulate, then scale by rs
    (void)hipMemsetAsync(hp, 0, SD * sizeof(float), stream);
    wmma_gemm_bf16<true, true><<<dim3(N_SUP / 64, 8), 256, 0, stream>>>(
        Q, xT, hp, N_PIX, N_SUP, N_PIX / 8);
    scale_rows_kernel<<<N_SUP, 256, 0, stream>>>(hp, rs);

    // 3) two MMPN layers
    const float* layer_in = hp;
    float* layer_out = h1;
    const float* Wl = W1; const float* bl = b1; const float* gl = g1; const float* bel = be1;
    for (int layer = 0; layer < 2; ++layer) {
        linear_kernel<<<N_SUP, 256, 0, stream>>>(layer_in, Wl, bl, lin);
        (void)hipMemsetAsync(agg, 0, SD * sizeof(float), stream);
        scatter_kernel<<<E_REG, 256, 0, stream>>>(es, ed, ew, lin, agg);
        scatter_kernel<<<E_IMP, 256, 0, stream>>>(is, id, iw, layer_in, agg);
        (void)hipMemsetAsync(gsum, 0, 2 * DD * sizeof(float), stream);
        bn_stats_kernel<<<N_SUP / 256, 256, 0, stream>>>(agg, gsum, gsumsq);
        bn_apply_kernel<<<N_SUP, 256, 0, stream>>>(agg, gsum, gsumsq, gl, bel, layer_out);
        layer_in = h1; layer_out = h2;
        Wl = W2; bl = b2; gl = g2; bel = be2;
    }

    // 4) h2T = bf16(h2^T);  out = Q @ h2
    transpose_to_bf16<<<dim3(N_SUP / 64, DD / 64), 256, 0, stream>>>(h2, h2T, N_SUP);
    wmma_gemm_bf16<false, false><<<dim3(N_PIX / 64, 1), 256, 0, stream>>>(
        Q, h2T, out, N_SUP, N_SUP, N_SUP);
}
